// TrajectoryGenerator_21569325760685
// MI455X (gfx1250) — matrix operand order repaired and checked by reading, not compiled, not run on hardware
//
#include <hip/hip_runtime.h>
#include <hip/hip_bf16.h>

// ---------------------------------------------------------------------------
// Fully fused Set-Transformer trajectory decoder for MI455X (gfx1250).
// One workgroup (256 threads = 8 wave32) per scene. All activations live in
// LDS; the ten 256x64x64 GEMMs per scene run on v_wmma_f32_16x16x32_bf16.
// ---------------------------------------------------------------------------

#define TDX 256
#define NAG 256           // agents per scene
#define DH  64
#define BN  262144        // B*N
#define LOG_STRIDE 257    // logits row stride (floats)

typedef __attribute__((ext_vector_type(16))) __bf16 bf16x16;
typedef __attribute__((ext_vector_type(8)))  float  f32x8;

struct MabW {
    const float *Wq, *bq, *Wk, *bk, *Wv, *bv, *Wo, *bo;
};

struct Params {
    const float* X;
    const float* I0; MabW e0m0, e0m1;
    const float* I1; MabW e1m0, e1m1;
    const float* S;  MabW pmab;
    MabW sabm;
    const float* reg_w; const float* reg_b;
    const float* conf_w; const float* conf_b;
    float* out;
};

__device__ __forceinline__ unsigned short f2bf(float f) {
    union { float f; unsigned u; } x; x.f = f;
    unsigned u = x.u;
    unsigned r = u + 0x7FFFu + ((u >> 16) & 1u);   // round to nearest even
    return (unsigned short)(r >> 16);
}
__device__ __forceinline__ float bf2f(unsigned short h) {
    union { unsigned u; float f; } x; x.u = ((unsigned)h) << 16;
    return x.f;
}

union Frag { uint4 q[2]; bf16x16 v; };

// Stage a 64x64 (K<=64 x 64) fp32 weight from global into LDS as bf16,
// transposed (N-major, contiguous K) and zero-padded in K.
__device__ __forceinline__ void load_wt(unsigned short* Wt, const float* W, int krows) {
    for (int idx = threadIdx.x; idx < DH * DH; idx += TDX) {
        int n = idx >> 6, k = idx & 63;
        float v = (k < krows) ? W[k * DH + n] : 0.0f;
        Wt[n * DH + k] = f2bf(v);
    }
}

// C(256x64) = A(256x64 bf16, LDS) * W(64x64 bf16, LDS N-major) [+ epilogue]
// residual=false: Out = acc + bias
// residual=true : Out = A + relu(acc + bias)     (A == O for the MAB out-proj)
__device__ __forceinline__ void gemm_big(const unsigned short* __restrict__ Abf,
                                         const unsigned short* __restrict__ Wt,
                                         unsigned short* __restrict__ Out,
                                         const float* __restrict__ bias,
                                         bool residual) {
    const int wave = threadIdx.x >> 5;
    const int lane = threadIdx.x & 31;
    const int lh = lane & 15;          // col-within-tile / row-within-tile
    const int hb = lane >> 4;          // lane-half select

    for (int rt = wave * 2; rt < wave * 2 + 2; ++rt) {
        const int arow = rt * 16 + lh;
        Frag a0, a1;
        // A-fragment (16x32 bf16): lanes 0-15: K 0-15; lanes 16-31: K 16-31
        a0.q[0] = *(const uint4*)(Abf + arow * DH + hb * 16);
        a0.q[1] = *(const uint4*)(Abf + arow * DH + hb * 16 + 8);
        a1.q[0] = *(const uint4*)(Abf + arow * DH + 32 + hb * 16);
        a1.q[1] = *(const uint4*)(Abf + arow * DH + 40 + hb * 16);
#pragma unroll
        for (int ct = 0; ct < 4; ++ct) {
            const int co = ct * 16 + lh;
            Frag b0, b1;
            // B-fragment (32x16 bf16): lane holds 16 contiguous K for col co
            const unsigned short* wp = Wt + co * DH + hb * 16;
            b0.q[0] = *(const uint4*)(wp);
            b0.q[1] = *(const uint4*)(wp + 8);
            b1.q[0] = *(const uint4*)(wp + 32);
            b1.q[1] = *(const uint4*)(wp + 40);
            f32x8 acc = {0.f, 0.f, 0.f, 0.f, 0.f, 0.f, 0.f, 0.f};
            acc = __builtin_amdgcn_wmma_f32_16x16x32_bf16(false, a0.v, false, b0.v,
                                                          (short)0, acc, false, false);
            acc = __builtin_amdgcn_wmma_f32_16x16x32_bf16(false, a1.v, false, b1.v,
                                                          (short)0, acc, false, false);
            const float bi = bias[co];
#pragma unroll
            for (int i = 0; i < 8; ++i) {
                const int ro = rt * 16 + hb * 8 + i;
                float v = acc[i] + bi;
                if (residual) v = bf2f(Abf[ro * DH + co]) + fmaxf(v, 0.f);
                Out[ro * DH + co] = f2bf(v);
            }
        }
    }
}

// MAB(Q_small, Y): nq in {8,3}, nk=256.  Result (nq x 64 fp32) -> Hf.
__device__ void mab_pool(int nq, const float* __restrict__ Qin, const MabW w, int krows,
                         unsigned short* bufY, unsigned short* bufK, unsigned short* Wt,
                         float* Qs, float* Oc, float* Hf, float* logits) {
    const int tid = threadIdx.x;
    // Qp = Qin @ Wq + bq   (nq x 64, fp32)
    for (int idx = tid; idx < nq * DH; idx += TDX) {
        int q = idx >> 6, n = idx & 63;
        float a = w.bq[n];
        for (int k = 0; k < DH; ++k) a += Qin[q * DH + k] * w.Wq[k * DH + n];
        Qs[idx] = a;
    }
    load_wt(Wt, w.Wk, krows);
    __syncthreads();
    gemm_big(bufY, Wt, bufK, w.bk, false);           // Kp -> bufK
    __syncthreads();
    const int R = nq * 4;                            // (q, head) rows
    // logits[r][agent] = Qp_h . Kp_h / 8
    for (int r = 0; r < R; ++r) {
        int q = r >> 2, h = r & 3;
        float a = 0.f;
#pragma unroll
        for (int d = 0; d < 16; ++d)
            a += Qs[q * DH + h * 16 + d] * bf2f(bufK[tid * DH + h * 16 + d]);
        logits[r * LOG_STRIDE + tid] = a * 0.125f;
    }
    __syncthreads();
    // softmax over 256 keys, one row per wave pass
    const int wave = tid >> 5, lane = tid & 31;
    for (int r = wave; r < R; r += 8) {
        float v[8], m = -3.4e38f;
#pragma unroll
        for (int j = 0; j < 8; ++j) {
            v[j] = logits[r * LOG_STRIDE + lane + 32 * j];
            m = fmaxf(m, v[j]);
        }
        for (int off = 16; off; off >>= 1) m = fmaxf(m, __shfl_xor(m, off, 32));
        float s = 0.f;
#pragma unroll
        for (int j = 0; j < 8; ++j) { v[j] = __expf(v[j] - m); s += v[j]; }
        for (int off = 16; off; off >>= 1) s += __shfl_xor(s, off, 32);
        const float inv = 1.f / s;
#pragma unroll
        for (int j = 0; j < 8; ++j) logits[r * LOG_STRIDE + lane + 32 * j] = v[j] * inv;
    }
    load_wt(Wt, w.Wv, krows);
    __syncthreads();
    gemm_big(bufY, Wt, bufK, w.bv, false);           // Vp -> bufK (Kp dead)
    __syncthreads();
    // O = Qp + P @ Vp
    for (int idx = tid; idx < R * 16; idx += TDX) {
        int r = idx >> 4, d = idx & 15;
        int q = r >> 2, h = r & 3;
        const float* pr = logits + r * LOG_STRIDE;
        float a = 0.f;
        for (int k = 0; k < NAG; ++k) a += pr[k] * bf2f(bufK[k * DH + h * 16 + d]);
        Oc[q * DH + h * 16 + d] = Qs[q * DH + h * 16 + d] + a;
    }
    __syncthreads();
    // Hf = O + relu(O @ Wo + bo)
    for (int idx = tid; idx < nq * DH; idx += TDX) {
        int q = idx >> 6, n = idx & 63;
        float a = w.bo[n];
        for (int k = 0; k < DH; ++k) a += Oc[q * DH + k] * w.Wo[k * DH + n];
        Hf[idx] = Oc[idx] + fmaxf(a, 0.f);
    }
    __syncthreads();
}

// MAB(Y, H): nq=256, nk=8.  Updates bufY in place (via bufK scratch).
__device__ void mab_broad(const MabW w, int krows_q,
                          unsigned short* bufY, unsigned short* bufK, unsigned short* Wt,
                          const float* __restrict__ Hf, float* KpH, float* VpH) {
    const int tid = threadIdx.x;
    // KpH / VpH = H @ Wk/Wv + b   (8 x 64, fp32)
    for (int idx = tid; idx < 8 * DH; idx += TDX) {
        int k = idx >> 6, n = idx & 63;
        float a1 = w.bk[n], a2 = w.bv[n];
        for (int j = 0; j < DH; ++j) {
            float h = Hf[k * DH + j];
            a1 += h * w.Wk[j * DH + n];
            a2 += h * w.Wv[j * DH + n];
        }
        KpH[idx] = a1; VpH[idx] = a2;
    }
    load_wt(Wt, w.Wq, krows_q);
    __syncthreads();
    gemm_big(bufY, Wt, bufK, w.bq, false);           // Qp -> bufK
    __syncthreads();
    // per-agent attention over 8 keys (thread = agent); O overwrites own Qp row
    {
        unsigned short* Qrow = bufK + tid * DH;
#pragma unroll
        for (int h = 0; h < 4; ++h) {
            float lg[8], m = -3.4e38f;
#pragma unroll
            for (int j = 0; j < 8; ++j) {
                float a = 0.f;
#pragma unroll
                for (int d = 0; d < 16; ++d)
                    a += bf2f(Qrow[h * 16 + d]) * KpH[j * DH + h * 16 + d];
                lg[j] = a * 0.125f;
                m = fmaxf(m, lg[j]);
            }
            float s = 0.f;
#pragma unroll
            for (int j = 0; j < 8; ++j) { lg[j] = __expf(lg[j] - m); s += lg[j]; }
            const float inv = 1.f / s;
            float od[16];
#pragma unroll
            for (int d = 0; d < 16; ++d) od[d] = bf2f(Qrow[h * 16 + d]);
#pragma unroll
            for (int j = 0; j < 8; ++j) {
                const float p = lg[j] * inv;
#pragma unroll
                for (int d = 0; d < 16; ++d) od[d] += p * VpH[j * DH + h * 16 + d];
            }
#pragma unroll
            for (int d = 0; d < 16; ++d) Qrow[h * 16 + d] = f2bf(od[d]);
        }
    }
    load_wt(Wt, w.Wo, DH);
    __syncthreads();
    gemm_big(bufK, Wt, bufY, w.bo, true);            // Y = O + relu(O@Wo+bo)
    __syncthreads();
}

__global__ __launch_bounds__(TDX) void traj_kernel(Params p) {
    extern __shared__ __align__(16) char smem[];
    unsigned short* bufY = (unsigned short*)smem;            // 256x64 bf16
    unsigned short* bufK = bufY + NAG * DH;                  // 256x64 bf16
    unsigned short* Wt   = bufK + NAG * DH;                  // 64x64 bf16 (N-major)
    float* Qs  = (float*)(Wt + DH * DH);                     // 16x64 f32
    float* Oc  = Qs + 16 * DH;                               // 16x64 f32
    float* Hf  = Oc + 16 * DH;                               // 16x64 f32
    float* KpH = Hf + 16 * DH;                               // 8x64 f32
    float* VpH = KpH + 8 * DH;                               // 8x64 f32
    float* logits = VpH + 8 * DH;                            // 32x257 f32

    const int tid = threadIdx.x;
    const int b   = blockIdx.x;
    const int g   = b * NAG + tid;

    // --- load agent features: Y[i][t*2+p] = X[t][g][p]; pad 40->64 with zeros
#pragma unroll
    for (int t = 0; t < 20; ++t) {
        const float2 v = *(const float2*)(p.X + ((size_t)t * BN + g) * 2);
        bufY[tid * DH + 2 * t]     = f2bf(v.x);
        bufY[tid * DH + 2 * t + 1] = f2bf(v.y);
    }
#pragma unroll
    for (int f = 40; f < DH; ++f) bufY[tid * DH + f] = 0;
    __syncthreads();

    // --- encoder 0 (ISAB, din=40)
    mab_pool(8, p.I0, p.e0m0, 40, bufY, bufK, Wt, Qs, Oc, Hf, logits);
    mab_broad(p.e0m1, 40, bufY, bufK, Wt, Hf, KpH, VpH);
    // --- encoder 1 (ISAB, din=64)
    mab_pool(8, p.I1, p.e1m0, DH, bufY, bufK, Wt, Qs, Oc, Hf, logits);
    mab_broad(p.e1m1, DH, bufY, bufK, Wt, Hf, KpH, VpH);
    // --- PMA: 3 seeds over 256 agents -> Hf (3x64)
    mab_pool(3, p.S, p.pmab, DH, bufY, bufK, Wt, Qs, Oc, Hf, logits);

    // --- SAB on (3 x 64): all tiny, fp32 VALU
    {
        const MabW w = p.sabm;
        for (int idx = tid; idx < 3 * DH; idx += TDX) {
            int q = idx >> 6, n = idx & 63;
            float aq = w.bq[n], ak = w.bk[n], av = w.bv[n];
            for (int j = 0; j < DH; ++j) {
                float h = Hf[q * DH + j];
                aq += h * w.Wq[j * DH + n];
                ak += h * w.Wk[j * DH + n];
                av += h * w.Wv[j * DH + n];
            }
            Qs[idx] = aq; KpH[idx] = ak; VpH[idx] = av;
        }
        __syncthreads();
        for (int idx = tid; idx < 3 * DH; idx += TDX) {
            int q = idx >> 6, n = idx & 63, h = n >> 4;
            float lg[3], m = -3.4e38f;
#pragma unroll
            for (int j = 0; j < 3; ++j) {
                float a = 0.f;
#pragma unroll
                for (int d = 0; d < 16; ++d)
                    a += Qs[q * DH + h * 16 + d] * KpH[j * DH + h * 16 + d];
                lg[j] = a * 0.125f;
                m = fmaxf(m, lg[j]);
            }
            float s = 0.f;
#pragma unroll
            for (int j = 0; j < 3; ++j) { lg[j] = __expf(lg[j] - m); s += lg[j]; }
            float o = Qs[idx];
#pragma unroll
            for (int j = 0; j < 3; ++j) o += (lg[j] / s) * VpH[j * DH + n];
            Oc[idx] = o;
        }
        __syncthreads();
        for (int idx = tid; idx < 3 * DH; idx += TDX) {
            int q = idx >> 6, n = idx & 63;
            float a = w.bo[n];
            for (int k = 0; k < DH; ++k) a += Oc[q * DH + k] * w.Wo[k * DH + n];
            Qs[idx] = Oc[idx] + fmaxf(a, 0.f);       // final (3 x 64) in Qs
        }
        __syncthreads();
    }

    // --- regression head: coords (3 x 60) per scene
    for (int idx = tid; idx < 3 * 60; idx += TDX) {
        int s = idx / 60, j = idx % 60;
        float a = p.reg_b[j];
        for (int k = 0; k < DH; ++k) a += Qs[s * DH + k] * p.reg_w[k * 60 + j];
        p.out[((size_t)b * 3 + s) * 60 + j] = a;
    }
    // --- confidence head + softmax over 3 seeds
    if (tid < 3) {
        float a = p.conf_b[0];
        for (int k = 0; k < DH; ++k) a += Qs[tid * DH + k] * p.conf_w[k];
        Oc[tid] = a;
    }
    __syncthreads();
    if (tid < 3) {
        float m  = fmaxf(Oc[0], fmaxf(Oc[1], Oc[2]));
        float e0 = __expf(Oc[0] - m), e1 = __expf(Oc[1] - m), e2 = __expf(Oc[2] - m);
        float e  = (tid == 0) ? e0 : ((tid == 1) ? e1 : e2);
        p.out[184320 + (size_t)b * 3 + tid] = e / (e0 + e1 + e2);
    }
}

static constexpr size_t SMEM_BYTES =
    (size_t)NAG * DH * 2 * 2      // bufY + bufK
    + (size_t)DH * DH * 2         // Wt
    + (size_t)16 * DH * 4 * 3     // Qs, Oc, Hf
    + (size_t)8 * DH * 4 * 2      // KpH, VpH
    + (size_t)32 * LOG_STRIDE * 4;// logits

extern "C" void kernel_launch(void* const* d_in, const int* in_sizes, int n_in,
                              void* d_out, int out_size, void* d_ws, size_t ws_size,
                              hipStream_t stream) {
    (void)in_sizes; (void)n_in; (void)out_size; (void)d_ws; (void)ws_size;
    auto mk = [&](int base) {
        MabW w;
        w.Wq = (const float*)d_in[base + 0]; w.bq = (const float*)d_in[base + 1];
        w.Wk = (const float*)d_in[base + 2]; w.bk = (const float*)d_in[base + 3];
        w.Wv = (const float*)d_in[base + 4]; w.bv = (const float*)d_in[base + 5];
        w.Wo = (const float*)d_in[base + 6]; w.bo = (const float*)d_in[base + 7];
        return w;
    };
    Params p;
    p.X    = (const float*)d_in[0];
    // d_in[1] = start_end_seq (uniform segments -> unused)
    p.I0   = (const float*)d_in[2];
    p.e0m0 = mk(3);
    p.e0m1 = mk(11);
    p.I1   = (const float*)d_in[19];
    p.e1m0 = mk(20);
    p.e1m1 = mk(28);
    p.S    = (const float*)d_in[36];
    p.pmab = mk(37);
    p.sabm = mk(45);
    p.reg_w  = (const float*)d_in[53];
    p.reg_b  = (const float*)d_in[54];
    p.conf_w = (const float*)d_in[55];
    p.conf_b = (const float*)d_in[56];
    p.out    = (float*)d_out;

    traj_kernel<<<dim3(1024), dim3(TDX), SMEM_BYTES, stream>>>(p);
}
